// lstm_74637941669960
// MI455X (gfx1250) — compile-verified, matrix operand order checked by reading, not run on hardware
//
#include <hip/hip_runtime.h>
#include <hip/hip_bf16.h>
#include <stdint.h>

// Problem dims
#define B_   64
#define S_   2048
#define F_   40
#define EMB_ 128
#define HID_ 256
#define G4_  1024          // 4*HID
#define R_   (B_ * S_)     // 131072 rows

typedef __bf16 v16bf __attribute__((ext_vector_type(16)));
typedef float  v8f   __attribute__((ext_vector_type(8)));
typedef unsigned int u32x4 __attribute__((ext_vector_type(4)));
typedef int          i32x4 __attribute__((ext_vector_type(4)));
typedef int          i32x8 __attribute__((ext_vector_type(8)));

__device__ __forceinline__ unsigned short f2bf(float f) {
  union { float f; unsigned u; } x; x.f = f;
  unsigned u = x.u;
  return (unsigned short)((u + 0x7FFFu + ((u >> 16) & 1u)) >> 16);  // RNE
}
__device__ __forceinline__ float bf2f(unsigned short h) {
  union { unsigned u; float f; } x; x.u = ((unsigned)h) << 16; return x.f;
}
__device__ __forceinline__ float sigm(float x) { return 1.0f / (1.0f + __expf(-x)); }

union Frag { unsigned u[8]; uint4 q[2]; v16bf v; };

// ---------------------------------------------------------------------------
// TDM: 2D tile load (bf16 elements) global -> LDS. One wave issues it.
// D# packed per CDNA5 ISA 8.3/8.4 (group0: count/lds/global/type; group1:
// data_size=2B, tensor_dim0/1, tile_dim0/1, dim0 stride).
// ---------------------------------------------------------------------------
__device__ __forceinline__ void tdm_load_2d_bf16(const void* gptr, unsigned lds_off_bytes,
                                                 unsigned rows, unsigned cols,
                                                 unsigned stride_elems) {
  unsigned long long ga = (unsigned long long)gptr;
  u32x4 g0;
  g0[0] = 1u;                                            // count=1, user desc
  g0[1] = lds_off_bytes;                                 // lds_addr
  g0[2] = (unsigned)(ga & 0xFFFFFFFFu);                  // global_addr lo
  g0[3] = ((unsigned)(ga >> 32) & 0x01FFFFFFu) | (2u << 30);  // addr hi | type=2
  i32x8 g1;
  g1[0] = (int)(1u << 16);                               // data_size=1 -> 2 bytes
  g1[1] = (int)((cols & 0xFFFFu) << 16);                 // tensor_dim0[15:0] @ bit48
  g1[2] = (int)((cols >> 16) | ((rows & 0xFFFFu) << 16));// dim0 hi | dim1 lo
  g1[3] = (int)((rows >> 16) | ((cols & 0xFFFFu) << 16));// dim1 hi | tile_dim0
  g1[4] = (int)(rows & 0xFFFFu);                         // tile_dim1 | tile_dim2=0
  g1[5] = (int)stride_elems;                             // tensor_dim0_stride lo
  g1[6] = 0; g1[7] = 0;
  i32x4 z4 = {0, 0, 0, 0};
#if __has_include(<hip/amd_detail/amd_gfx1250_TDM.h>)
  i32x8 z8 = {0, 0, 0, 0, 0, 0, 0, 0};
  __builtin_amdgcn_tensor_load_to_lds(g0, g1, z4, z4, z8, 0);
#else
  __builtin_amdgcn_tensor_load_to_lds(g0, g1, z4, z4, 0);
#endif
}

// ---------------------------------------------------------------------------
// Prep: convert weights to bf16 (W_ih plain; W_hh permuted so each workgroup's
// 256 gate rows are contiguous: [q=0..3][g*64+j][k], orig row g*256+q*64+j),
// combine biases, zero h double-buffers and barrier counters.
// ---------------------------------------------------------------------------
__global__ void prep_kernel(const float* Wih0, const float* Whh0, const float* bih0, const float* bhh0,
                            const float* Wih1, const float* Whh1, const float* bih1, const float* bhh1,
                            unsigned short* wih0b, unsigned short* wih1b,
                            unsigned short* whh0p, unsigned short* whh1p,
                            float* b0g, float* b1g, unsigned short* hb, unsigned* ctr) {
  long idx = (long)blockIdx.x * 256 + threadIdx.x;
  long n;
  n = 1024L * 128; if (idx < n) { wih0b[idx] = f2bf(Wih0[idx]); return; } idx -= n;
  n = 1024L * 256; if (idx < n) { wih1b[idx] = f2bf(Wih1[idx]); return; } idx -= n;
  n = 1024L * 256; if (idx < n) {
    long k = idx & 255, rp = idx >> 8;
    long q = rp >> 8, r = rp & 255, g = r >> 6, j = r & 63;
    whh0p[idx] = f2bf(Whh0[(g * 256 + q * 64 + j) * 256 + k]); return; } idx -= n;
  n = 1024L * 256; if (idx < n) {
    long k = idx & 255, rp = idx >> 8;
    long q = rp >> 8, r = rp & 255, g = r >> 6, j = r & 63;
    whh1p[idx] = f2bf(Whh1[(g * 256 + q * 64 + j) * 256 + k]); return; } idx -= n;
  n = 1024; if (idx < n) { b0g[idx] = bih0[idx] + bhh0[idx]; return; } idx -= n;
  n = 1024; if (idx < n) { b1g[idx] = bih1[idx] + bhh1[idx]; return; } idx -= n;
  n = 4L * 64 * 256; if (idx < n) { hb[idx] = 0; return; } idx -= n;
  if (idx < 8) ctr[idx] = 0;
}

// ---------------------------------------------------------------------------
// Input projection: x = in_states @ W_inT + b_in  (K=40, 0.5% of FLOPs; VALU)
// ---------------------------------------------------------------------------
__global__ void inproj_kernel(const float* xin, const float* Win, const float* bin,
                              unsigned short* xbf) {
  long idx = (long)blockIdx.x * 256 + threadIdx.x;     // R_*EMB_
  int  e = (int)(idx & (EMB_ - 1));
  long row = idx >> 7;
  const float* xr = xin + row * F_;
  const float* wr = Win + (long)e * F_;
  float s = bin[e];
#pragma unroll
  for (int k = 0; k < F_; ++k) s = fmaf(xr[k], wr[k], s);
  xbf[idx] = f2bf(s);
}

// ---------------------------------------------------------------------------
// gx = X(bf16,[R,K]) @ W(bf16,[1024,K])^T + bias -> bf16 [R,1024]
// One wave per 16x64 output tile; 4 WMMA tiles per wave, K/32 k-steps.
// ---------------------------------------------------------------------------
__global__ void gx_gemm_kernel(const unsigned short* __restrict__ X,
                               const unsigned short* __restrict__ W,
                               const float* __restrict__ bias,
                               unsigned short* __restrict__ OutG, int K) {
  int lane = threadIdx.x & 31, wv = threadIdx.x >> 5;
  int wg = blockIdx.x * 8 + wv;
  int ntile = wg & 15;                 // 16 tiles of 64 cols
  long m0 = (long)(wg >> 4) * 16;
  int r = lane & 15, half = lane >> 4;

  v8f z = {0, 0, 0, 0, 0, 0, 0, 0};
  v8f acc[4] = {z, z, z, z};

  const unsigned short* arow = X + (m0 + r) * K;
  int nk = K >> 5;
  for (int ks = 0; ks < nk; ++ks) {
    int kb = ks * 32 + half * 8;
    Frag a;
    a.q[0] = *(const uint4*)(arow + kb);
    a.q[1] = *(const uint4*)(arow + kb + 16);
#pragma unroll
    for (int g = 0; g < 4; ++g) {
      int nn = ntile * 64 + g * 16 + r;
      const unsigned short* brow = W + (long)nn * K;
      Frag b;
      b.q[0] = *(const uint4*)(brow + kb);
      b.q[1] = *(const uint4*)(brow + kb + 16);
      acc[g] = __builtin_amdgcn_wmma_f32_16x16x32_bf16(false, a.v, false, b.v,
                                                       (short)0, acc[g], false, false);
    }
  }
#pragma unroll
  for (int g = 0; g < 4; ++g) {
    int nn = ntile * 64 + g * 16 + r;
    float bv = bias[nn];
#pragma unroll
    for (int i = 0; i < 8; ++i) {
      long rr = m0 + half * 8 + i;
      OutG[rr * G4_ + nn] = f2bf(acc[g][i] + bv);
    }
  }
}

// ---------------------------------------------------------------------------
// Recurrent scan. Grid = 16 WGs: (batch tile bt 0..3) x (H-chunk q 0..3).
// W_hh slice (256x256 bf16 = 128KB) TDM-loaded into LDS once. Per step:
// stage full h tile [16x256] LDS <- global double buffer, 4 waves each do
// their i/f/g/o 16x16 tiles (32 WMMAs/wave/step), f32 gate math, write h/y,
// then barrier across the 4 WGs of this batch tile (atomic + cluster barrier).
// ---------------------------------------------------------------------------
__global__ void lstm_rec_kernel(const unsigned short* __restrict__ gx,
                                const unsigned short* __restrict__ WhhPerm,
                                unsigned short* __restrict__ hb,   // [2][64*256]
                                unsigned short* __restrict__ y,    // [R_,256]
                                unsigned* __restrict__ ctr) {      // [4]
  extern __shared__ unsigned short smem[];
  unsigned short* sW = smem;                // 256*256 bf16 = 128KB
  unsigned short* sH = smem + 256 * 256;    // 16*256  bf16 = 8KB

  const int tid = threadIdx.x, lane = tid & 31, wv = tid >> 5;
  const int bt = blockIdx.x >> 2, q = blockIdx.x & 3;
  const int b0 = bt * 16, hc0 = q * 64;
  const int r = lane & 15, half = lane >> 4;

  if (wv == 0) {
    tdm_load_2d_bf16(WhhPerm + (long)q * 256 * 256, 0u, 256u, 256u, 256u);
    __builtin_amdgcn_s_wait_tensorcnt(0);
  }
  __syncthreads();

  v8f z = {0, 0, 0, 0, 0, 0, 0, 0};
  v8f cst = z;                              // cell state, f32, wave-local tile

  for (int t = 0; t < S_; ++t) {
    // stage h_t tile into LDS (8KB, 128 threads x 4 x b128)
    {
      const uint4* src = (const uint4*)(hb + (long)(t & 1) * (64 * 256) + b0 * 256);
      uint4* dst = (uint4*)sH;
      for (int j = tid; j < 512; j += 128) dst[j] = src[j];
    }
    __syncthreads();

    // accumulators init from gx slice (bf16 -> f32)
    v8f acc[4];
#pragma unroll
    for (int g = 0; g < 4; ++g) {
      int col = g * 256 + hc0 + wv * 16 + r;
#pragma unroll
      for (int i = 0; i < 8; ++i) {
        long rr = (long)(b0 + half * 8 + i) * S_ + t;
        acc[g][i] = bf2f(gx[rr * G4_ + col]);
      }
    }

    // h_t @ W_hh^T : K = 256 -> 8 k-steps of 16x16x32
#pragma unroll
    for (int ks = 0; ks < 8; ++ks) {
      int kb = ks * 32 + half * 8;
      Frag a;
      a.q[0] = *(const uint4*)(sH + r * 256 + kb);
      a.q[1] = *(const uint4*)(sH + r * 256 + kb + 16);
#pragma unroll
      for (int g = 0; g < 4; ++g) {
        int nn = g * 64 + wv * 16 + r;     // permuted gate row
        Frag b;
        b.q[0] = *(const uint4*)(sW + nn * 256 + kb);
        b.q[1] = *(const uint4*)(sW + nn * 256 + kb + 16);
        acc[g] = __builtin_amdgcn_wmma_f32_16x16x32_bf16(false, a.v, false, b.v,
                                                         (short)0, acc[g], false, false);
      }
    }

    // gates + state update (i,f,g,o), write h_{t+1} and y_t
    unsigned short* hdst = hb + (long)((t + 1) & 1) * (64 * 256);
    int col = hc0 + wv * 16 + r;
#pragma unroll
    for (int i = 0; i < 8; ++i) {
      float iv = sigm(acc[0][i]);
      float fv = sigm(acc[1][i]);
      float gv = tanhf(acc[2][i]);
      float ov = sigm(acc[3][i]);
      float c = fv * cst[i] + iv * gv;
      cst[i] = c;
      unsigned short hv = f2bf(ov * tanhf(c));
      int rowm = b0 + half * 8 + i;
      hdst[rowm * 256 + col] = hv;
      y[((long)rowm * S_ + t) * 256 + col] = hv;
    }

    // make h visible device-wide, then sync the 4 WGs of this batch tile
    __threadfence();
    __syncthreads();
    if (tid == 0) {
      atomicAdd(&ctr[bt], 1u);
      unsigned tgt = 4u * (unsigned)(t + 1);
      while (__hip_atomic_load(&ctr[bt], __ATOMIC_ACQUIRE, __HIP_MEMORY_SCOPE_AGENT) < tgt)
        __builtin_amdgcn_s_sleep(2);
    }
    __builtin_amdgcn_s_cluster_barrier();   // CDNA5 cluster barrier (NOP if not clustered)
    __syncthreads();
  }
}

// ---------------------------------------------------------------------------
// out[b,s] = y1[b,s,:] . W_out + b_out   (wave-per-row reduction)
// ---------------------------------------------------------------------------
__global__ void outproj_kernel(const unsigned short* __restrict__ y1,
                               const float* __restrict__ Wout,
                               const float* __restrict__ bout,
                               float* __restrict__ out) {
  int lane = threadIdx.x & 31, wv = threadIdx.x >> 5;
  long row = (long)blockIdx.x * 8 + wv;
  const unsigned short* yr = y1 + row * 256;
  float s = 0.0f;
#pragma unroll
  for (int j = 0; j < 8; ++j) {
    int h = j * 32 + lane;
    s = fmaf(bf2f(yr[h]), Wout[h], s);
  }
#pragma unroll
  for (int off = 16; off; off >>= 1) s += __shfl_xor(s, off, 32);
  if (lane == 0) out[row] = s + bout[0];
}

// ---------------------------------------------------------------------------
extern "C" void kernel_launch(void* const* d_in, const int* in_sizes, int n_in,
                              void* d_out, int out_size, void* d_ws, size_t ws_size,
                              hipStream_t stream) {
  (void)in_sizes; (void)n_in; (void)out_size; (void)ws_size;
  const float* in_states = (const float*)d_in[0];
  const float* W_in  = (const float*)d_in[1];
  const float* b_in  = (const float*)d_in[2];
  const float* W_ih0 = (const float*)d_in[3];
  const float* W_hh0 = (const float*)d_in[4];
  const float* b_ih0 = (const float*)d_in[5];
  const float* b_hh0 = (const float*)d_in[6];
  const float* W_ih1 = (const float*)d_in[7];
  const float* W_hh1 = (const float*)d_in[8];
  const float* b_ih1 = (const float*)d_in[9];
  const float* b_hh1 = (const float*)d_in[10];
  const float* W_out = (const float*)d_in[11];
  const float* b_out = (const float*)d_in[12];
  float* out = (float*)d_out;

  // workspace layout (256B aligned)
  char* ws = (char*)d_ws;
  size_t o = 0;
  auto take = [&](size_t bytes) { char* p = ws + o; o += (bytes + 255) & ~(size_t)255; return p; };
  unsigned short* wih0b = (unsigned short*)take(1024L * 128 * 2);
  unsigned short* wih1b = (unsigned short*)take(1024L * 256 * 2);
  unsigned short* whh0p = (unsigned short*)take(1024L * 256 * 2);
  unsigned short* whh1p = (unsigned short*)take(1024L * 256 * 2);
  float*          b0g   = (float*)take(1024 * 4);
  float*          b1g   = (float*)take(1024 * 4);
  unsigned*       ctr   = (unsigned*)take(8 * 4);
  unsigned short* hbAll = (unsigned short*)take(4L * 64 * 256 * 2);   // 2 dbl-bufs
  unsigned short* xbf   = (unsigned short*)take((size_t)R_ * EMB_ * 2);
  unsigned short* y0    = (unsigned short*)take((size_t)R_ * HID_ * 2);
  unsigned short* y1    = (unsigned short*)take((size_t)R_ * HID_ * 2);
  unsigned short* gxb   = (unsigned short*)take((size_t)R_ * G4_ * 2);

  // 1) weight prep (also zeroes h buffers + barrier counters each launch)
  {
    long total = 1024L * 128 + 3 * 1024L * 256 + 2 * 1024 + 4L * 64 * 256 + 8;
    int blocks = (int)((total + 255) / 256);
    prep_kernel<<<blocks, 256, 0, stream>>>(W_ih0, W_hh0, b_ih0, b_hh0,
                                            W_ih1, W_hh1, b_ih1, b_hh1,
                                            wih0b, wih1b, whh0p, whh1p,
                                            b0g, b1g, hbAll, ctr);
  }
  // 2) input projection -> x bf16 [R,128]
  inproj_kernel<<<(R_ * EMB_) / 256, 256, 0, stream>>>(in_states, W_in, b_in, xbf);

  const size_t recSmem = (256 * 256 + 16 * 256) * sizeof(unsigned short);  // 136KB

  // 3) layer 0: gx then scan
  gx_gemm_kernel<<<(R_ / 16) * 16 / 8, 256, 0, stream>>>(xbf, wih0b, b0g, gxb, EMB_);
  lstm_rec_kernel<<<16, 128, recSmem, stream>>>(gxb, whh0p, hbAll, y0, ctr);

  // 4) layer 1: gx then scan (reuse gx buffer; second h dbl-buffer + counters)
  gx_gemm_kernel<<<(R_ / 16) * 16 / 8, 256, 0, stream>>>(y0, wih1b, b1g, gxb, HID_);
  lstm_rec_kernel<<<16, 128, recSmem, stream>>>(gxb, whh1p, hbAll + 2L * 64 * 256, y1, ctr + 4);

  // 5) output projection
  outproj_kernel<<<R_ / 8, 256, 0, stream>>>(y1, W_out, b_out, out);
}
